// FineGrainedFP8SwiGLUMLP_63342177681984
// MI455X (gfx1250) — 8-bit matrix operand order repaired and checked by reading, compiled, not run on hardware
//
#include <hip/hip_runtime.h>
#include <hip/hip_bf16.h>

typedef int   v16i __attribute__((ext_vector_type(16)));
typedef float v8f  __attribute__((ext_vector_type(8)));

#define FP8_MAX 448.0f
#define ROWP 144

#if __has_builtin(__builtin_amdgcn_global_load_async_to_lds_b128) && \
    __has_builtin(__builtin_amdgcn_s_wait_asynccnt)
#define USE_ASYNC_LDS 1
#endif

typedef int vec4i __attribute__((vector_size(16)));
typedef __attribute__((address_space(1))) vec4i gbl_vec4i;
typedef __attribute__((address_space(3))) vec4i lds_vec4i;

__device__ __forceinline__ void cp16_async(const unsigned char* g, unsigned char* l) {
#if defined(USE_ASYNC_LDS)
    __builtin_amdgcn_global_load_async_to_lds_b128(
        (gbl_vec4i*)g, (lds_vec4i*)l, 0, 0);
#else
    *reinterpret_cast<int4*>(l) = *reinterpret_cast<const int4*>(g);
#endif
}

__device__ __forceinline__ void async_wait_all() {
#if defined(USE_ASYNC_LDS)
    __builtin_amdgcn_s_wait_asynccnt(0);
#endif
}

__device__ __forceinline__ unsigned char fp8_e4m3(float x) {
    x = fminf(fmaxf(x, -FP8_MAX), FP8_MAX);
#if __has_builtin(__builtin_amdgcn_cvt_pk_fp8_f32)
    int p = __builtin_amdgcn_cvt_pk_fp8_f32(x, 0.0f, 0, false);
    return (unsigned char)(p & 0xff);
#else
    unsigned int u = __float_as_uint(x);
    unsigned char sign = (unsigned char)((u >> 31) << 7);
    float ax = fabsf(x);
    if (ax < 9.765625e-04f) return sign;
    int ee = (int)((__float_as_uint(ax) >> 23) & 0xff) - 127;
    unsigned char body;
    if (ee < -6) {
        body = (unsigned char)(int)rintf(ax * 512.0f);
    } else {
        float mant = ax * exp2f((float)(-ee));
        int mi = (int)rintf((mant - 1.0f) * 8.0f);
        if (mi == 8) { mi = 0; ++ee; }
        if (ee > 8)  { ee = 8; mi = 6; }
        body = (unsigned char)(((ee + 7) << 3) | mi);
    }
    return sign | body;
#endif
}

__device__ __forceinline__ v16i load_a_frag(const unsigned char* __restrict__ rowp,
                                            int laneHalf) {
    v16i a;
#pragma unroll
    for (int p = 0; p < 8; ++p) {
        int k = (p >> 1) * 32 + laneHalf * 16 + (p & 1) * 8;
        int2 d = *reinterpret_cast<const int2*>(rowp + k);
        a[2 * p]     = d.x;
        a[2 * p + 1] = d.y;
    }
    return a;
}

__device__ __forceinline__ v16i load_b_frag(const unsigned char* __restrict__ rowp,
                                            int laneHalf) {
    v16i b;
#pragma unroll
    for (int g = 0; g < 4; ++g) {
        int4 d = *reinterpret_cast<const int4*>(rowp + g * 32 + laneHalf * 16);
        b[4 * g]     = d.x;
        b[4 * g + 1] = d.y;
        b[4 * g + 2] = d.z;
        b[4 * g + 3] = d.w;
    }
    return b;
}

__global__ __launch_bounds__(256) void k_quant_x(const float* __restrict__ x,
                                                 unsigned char* __restrict__ xq,
                                                 float* __restrict__ xs,
                                                 int ngroups) {
    int lane = threadIdx.x & 31;
    int wave = threadIdx.x >> 5;
    int g = blockIdx.x * 8 + wave;
    if (g >= ngroups) return;
    const float4 v = *reinterpret_cast<const float4*>(x + (size_t)g * 128 + lane * 4);
    float amax = fmaxf(fmaxf(fabsf(v.x), fabsf(v.y)), fmaxf(fabsf(v.z), fabsf(v.w)));
#pragma unroll
    for (int m = 16; m >= 1; m >>= 1) amax = fmaxf(amax, __shfl_xor(amax, m, 32));
    float scale = fmaxf(amax * (1.0f / FP8_MAX), 1e-12f);
    float r = 1.0f / scale;
    unsigned int packed = (unsigned int)fp8_e4m3(v.x * r)
                        | ((unsigned int)fp8_e4m3(v.y * r) << 8)
                        | ((unsigned int)fp8_e4m3(v.z * r) << 16)
                        | ((unsigned int)fp8_e4m3(v.w * r) << 24);
    *reinterpret_cast<unsigned int*>(xq + (size_t)g * 128 + lane * 4) = packed;
    if (lane == 0) xs[g] = scale;
}

__global__ __launch_bounds__(256) void k_gate_up(
    const unsigned char* __restrict__ xq, const float* __restrict__ xs,
    const unsigned char* __restrict__ gw, const float* __restrict__ gsc,
    const unsigned char* __restrict__ uw, const float* __restrict__ usc,
    unsigned char* __restrict__ hq, float* __restrict__ hs,
    int K, int N) {
    const int KB = K >> 7;
    const int NB = N >> 7;
    const int tid = threadIdx.x;
    const int lane = tid & 31;
    const int wave = tid >> 5;
    const int wm = wave >> 1;
    const int wn = wave & 1;
    const int laneHalf = lane >> 4;
    const int lane16 = lane & 15;
    const int row0 = blockIdx.x * 128;
    const int nblk = blockIdx.y;
    const int col0 = nblk * 128;

    extern __shared__ __align__(16) unsigned char smem[];
    unsigned char* sA = smem;
    unsigned char* sG = sA + 2 * 128 * ROWP;
    unsigned char* sU = sG + 2 * 128 * ROWP;
    __shared__ float s_scl[2][128];
    __shared__ float s_part[2][128];
    __shared__ float s_scale[128];

    auto stage = [&](int buf, int kb) {
        const int koff = kb << 7;
#pragma unroll
        for (int t = 0; t < 4; ++t) {
            const int idx = tid + (t << 8);
            const int r = idx >> 3;
            const int c = (idx & 7) << 4;
            const size_t loff = (size_t)(buf * 128 + r) * ROWP + c;
            cp16_async(xq + (size_t)(row0 + r) * K + (koff + c), sA + loff);
            const size_t goff = (size_t)(col0 + r) * K + (koff + c);
            cp16_async(gw + goff, sG + loff);
            cp16_async(uw + goff, sU + loff);
        }
        if (tid < 128) s_scl[buf][tid] = xs[(size_t)(row0 + tid) * KB + kb];
    };

    stage(0, 0);

    v8f accg[2][4], accu[2][4];
#pragma unroll
    for (int a = 0; a < 2; ++a)
#pragma unroll
        for (int b = 0; b < 4; ++b) { v8f z = {}; accg[a][b] = z; accu[a][b] = z; }

    const float* gscRow = gsc + (size_t)nblk * KB;
    const float* uscRow = usc + (size_t)nblk * KB;

    for (int kb = 0; kb < KB; ++kb) {
        const int cur = kb & 1;
        async_wait_all();
        __syncthreads();
        if (kb + 1 < KB) stage(cur ^ 1, kb + 1);

        v16i afr[2];
#pragma unroll
        for (int tm = 0; tm < 2; ++tm)
            afr[tm] = load_a_frag(
                sA + (size_t)(cur * 128 + wm * 32 + tm * 16 + lane16) * ROWP, laneHalf);
        float asv[2][8];
#pragma unroll
        for (int tm = 0; tm < 2; ++tm)
#pragma unroll
            for (int j = 0; j < 8; ++j)
                asv[tm][j] = s_scl[cur][wm * 32 + tm * 16 + laneHalf * 8 + j];
        const float bg = gscRow[kb];
        const float bu = uscRow[kb];
#pragma unroll
        for (int tn = 0; tn < 4; ++tn) {
            const size_t rB = (size_t)(cur * 128 + wn * 64 + tn * 16 + lane16) * ROWP;
            v16i bfg = load_b_frag(sG + rB, laneHalf);
            v16i bfu = load_b_frag(sU + rB, laneHalf);
#pragma unroll
            for (int tm = 0; tm < 2; ++tm) {
                v8f z = {};
                v8f cg = __builtin_amdgcn_wmma_f32_16x16x128_fp8_fp8(
                    afr[tm], bfg, (short)0, z, false, false);
                v8f cu = __builtin_amdgcn_wmma_f32_16x16x128_fp8_fp8(
                    afr[tm], bfu, (short)0, z, false, false);
#pragma unroll
                for (int j = 0; j < 8; ++j) {
                    accg[tm][tn][j] += cg[j] * (asv[tm][j] * bg);
                    accu[tm][tn][j] += cu[j] * (asv[tm][j] * bu);
                }
            }
        }
    }

    float hv[2][4][8];
    float rmax[2][8];
#pragma unroll
    for (int tm = 0; tm < 2; ++tm)
#pragma unroll
        for (int j = 0; j < 8; ++j) rmax[tm][j] = 0.0f;
#pragma unroll
    for (int tm = 0; tm < 2; ++tm)
#pragma unroll
        for (int tn = 0; tn < 4; ++tn)
#pragma unroll
            for (int j = 0; j < 8; ++j) {
                float g = accg[tm][tn][j];
                float u = accu[tm][tn][j];
                float h = (g / (1.0f + __expf(-g))) * u;
                hv[tm][tn][j] = h;
                rmax[tm][j] = fmaxf(rmax[tm][j], fabsf(h));
            }
#pragma unroll
    for (int tm = 0; tm < 2; ++tm)
#pragma unroll
        for (int j = 0; j < 8; ++j) {
            float v = rmax[tm][j];
            v = fmaxf(v, __shfl_xor(v, 1, 32));
            v = fmaxf(v, __shfl_xor(v, 2, 32));
            v = fmaxf(v, __shfl_xor(v, 4, 32));
            v = fmaxf(v, __shfl_xor(v, 8, 32));
            rmax[tm][j] = v;
        }
    if (lane16 == 0) {
#pragma unroll
        for (int tm = 0; tm < 2; ++tm)
#pragma unroll
            for (int j = 0; j < 8; ++j)
                s_part[wn][wm * 32 + tm * 16 + laneHalf * 8 + j] = rmax[tm][j];
    }
    __syncthreads();
    if (tid < 128) {
        float v = fmaxf(s_part[0][tid], s_part[1][tid]);
        float sc = fmaxf(v * (1.0f / FP8_MAX), 1e-12f);
        s_scale[tid] = sc;
        hs[(size_t)(row0 + tid) * NB + nblk] = sc;
    }
    __syncthreads();
#pragma unroll
    for (int tm = 0; tm < 2; ++tm)
#pragma unroll
        for (int j = 0; j < 8; ++j) {
            int mloc = wm * 32 + tm * 16 + laneHalf * 8 + j;
            float rcp = 1.0f / s_scale[mloc];
            size_t mrow = (size_t)(row0 + mloc) * N;
#pragma unroll
            for (int tn = 0; tn < 4; ++tn) {
                int n = col0 + wn * 64 + tn * 16 + lane16;
                hq[mrow + n] = fp8_e4m3(hv[tm][tn][j] * rcp);
            }
        }
}

__global__ __launch_bounds__(256) void k_down(
    const unsigned char* __restrict__ aq, const float* __restrict__ asc,
    const unsigned char* __restrict__ w,  const float* __restrict__ wsc,
    float* __restrict__ out, int K, int N) {
    const int KB = K >> 7;
    const int tid = threadIdx.x;
    const int lane = tid & 31;
    const int wave = tid >> 5;
    const int wm = wave >> 1;
    const int wn = wave & 1;
    const int laneHalf = lane >> 4;
    const int lane16 = lane & 15;
    const int row0 = blockIdx.x * 128;
    const int col0 = blockIdx.y * 128;

    extern __shared__ __align__(16) unsigned char smem[];
    unsigned char* sA = smem;
    unsigned char* sB = sA + 2 * 128 * ROWP;
    __shared__ float s_scl[2][128];

    auto stage = [&](int buf, int kb) {
        const int koff = kb << 7;
#pragma unroll
        for (int t = 0; t < 4; ++t) {
            const int idx = tid + (t << 8);
            const int r = idx >> 3;
            const int c = (idx & 7) << 4;
            const size_t loff = (size_t)(buf * 128 + r) * ROWP + c;
            cp16_async(aq + (size_t)(row0 + r) * K + (koff + c), sA + loff);
            cp16_async(w  + (size_t)(col0 + r) * K + (koff + c), sB + loff);
        }
        if (tid < 128) s_scl[buf][tid] = asc[(size_t)(row0 + tid) * KB + kb];
    };

    stage(0, 0);

    v8f acc[2][4];
#pragma unroll
    for (int a = 0; a < 2; ++a)
#pragma unroll
        for (int b = 0; b < 4; ++b) { v8f z = {}; acc[a][b] = z; }

    const float* wscRow = wsc + (size_t)blockIdx.y * KB;

    for (int kb = 0; kb < KB; ++kb) {
        const int cur = kb & 1;
        async_wait_all();
        __syncthreads();
        if (kb + 1 < KB) stage(cur ^ 1, kb + 1);

        v16i afr[2];
#pragma unroll
        for (int tm = 0; tm < 2; ++tm)
            afr[tm] = load_a_frag(
                sA + (size_t)(cur * 128 + wm * 32 + tm * 16 + lane16) * ROWP, laneHalf);
        float asv[2][8];
#pragma unroll
        for (int tm = 0; tm < 2; ++tm)
#pragma unroll
            for (int j = 0; j < 8; ++j)
                asv[tm][j] = s_scl[cur][wm * 32 + tm * 16 + laneHalf * 8 + j];
        const float bs = wscRow[kb];
#pragma unroll
        for (int tn = 0; tn < 4; ++tn) {
            const size_t rB = (size_t)(cur * 128 + wn * 64 + tn * 16 + lane16) * ROWP;
            v16i bf = load_b_frag(sB + rB, laneHalf);
#pragma unroll
            for (int tm = 0; tm < 2; ++tm) {
                v8f z = {};
                v8f c = __builtin_amdgcn_wmma_f32_16x16x128_fp8_fp8(
                    afr[tm], bf, (short)0, z, false, false);
#pragma unroll
                for (int j = 0; j < 8; ++j)
                    acc[tm][tn][j] += c[j] * (asv[tm][j] * bs);
            }
        }
    }

#pragma unroll
    for (int tm = 0; tm < 2; ++tm)
#pragma unroll
        for (int j = 0; j < 8; ++j) {
            int mloc = wm * 32 + tm * 16 + laneHalf * 8 + j;
            size_t orow = (size_t)(row0 + mloc) * N;
#pragma unroll
            for (int tn = 0; tn < 4; ++tn) {
                int n = col0 + wn * 64 + tn * 16 + lane16;
                out[orow + n] = acc[tm][tn][j];
            }
        }
}

extern "C" void kernel_launch(void* const* d_in, const int* in_sizes, int n_in,
                              void* d_out, int out_size, void* d_ws, size_t ws_size,
                              hipStream_t stream) {
    const float*         x   = (const float*)d_in[0];
    const unsigned char* gw  = (const unsigned char*)d_in[1];
    const float*         gsc = (const float*)d_in[2];
    const unsigned char* uw  = (const unsigned char*)d_in[3];
    const float*         usc = (const float*)d_in[4];
    const unsigned char* dw  = (const unsigned char*)d_in[5];
    const float*         dsc = (const float*)d_in[6];
    float*               out = (float*)d_out;

    const int M = 4096, H = 4096, I = 14336;

    unsigned char* ws = (unsigned char*)d_ws;
    size_t off = 0;
    unsigned char* xq = ws + off;            off += (size_t)M * H;
    float*         xs = (float*)(ws + off);  off += (size_t)M * (H / 128) * sizeof(float);
    unsigned char* hq = ws + off;            off += (size_t)M * I;
    float*         hs = (float*)(ws + off);

    const int ngroups = M * (H / 128);
    k_quant_x<<<ngroups / 8, 256, 0, stream>>>(x, xq, xs, ngroups);

    const size_t shGU = (size_t)3 * 2 * 128 * ROWP;
    dim3 g1(M / 128, I / 128);
    k_gate_up<<<g1, 256, shGU, stream>>>(xq, xs, gw, gsc, uw, usc, hq, hs, H, I);

    const size_t shDN = (size_t)2 * 2 * 128 * ROWP;
    dim3 g2(M / 128, H / 128);
    k_down<<<g2, 256, shDN, stream>>>(hq, hs, dw, dsc, out, I, H);
}
